// LinearAttention_29463475651327
// MI455X (gfx1250) — matrix operand order repaired and checked by reading, not compiled, not run on hardware
//
#include <hip/hip_runtime.h>
#include <hip/hip_bf16.h>

typedef __attribute__((ext_vector_type(16))) __bf16 v16bf;
typedef __attribute__((ext_vector_type(8)))  float  v8f;

#define DMODEL 1024
#define NHEADS 16
#define HDIM   64
#define SEQ    4096
#define BATCH  4
#define ROWS   (BATCH * SEQ) /* 16384 */
#define EPSV   1e-6f

// ---- CDNA5 async global->LDS via inline asm (portable across toolchains) ----
// VDST = VGPR with LDS byte address; VADDR = 64-bit global address; SADDR=off.
__device__ __forceinline__ void async_copy_b128(unsigned lds_off, const void* gaddr) {
  asm volatile("global_load_async_to_lds_b128 %0, %1, off"
               :
               : "v"(lds_off), "v"((unsigned long long)(size_t)gaddr)
               : "memory");
}

#if defined(__has_builtin) && __has_builtin(__builtin_amdgcn_s_wait_asynccnt)
#define WAIT_ASYNC(n) __builtin_amdgcn_s_wait_asynccnt(n)
#else
#define WAIT_ASYNC(n) asm volatile("s_wait_asynccnt %0" ::"i"(n) : "memory")
#endif

__device__ __forceinline__ unsigned short f2bf(float f) {
  unsigned u = __builtin_bit_cast(unsigned, f);
  u += 0x7FFFu + ((u >> 16) & 1u);  // round-to-nearest-even
  return (unsigned short)(u >> 16);
}
__device__ __forceinline__ float bf2f(unsigned short h) {
  unsigned u = ((unsigned)h) << 16;
  return __builtin_bit_cast(float, u);
}

union Frag { v16bf v; uint4 u[2]; unsigned short s[16]; };

__device__ __forceinline__ v8f wmma_bf16(const Frag& a, const Frag& b, v8f c) {
  return __builtin_amdgcn_wmma_f32_16x16x32_bf16(false, a.v, false, b.v, (short)0, c,
                                                 false, false);
}

enum { ACT_NONE = 0, ACT_PHI = 1 };
enum { OUT_Q = 0, OUT_TR = 1, OUT_FINAL = 2 };

// ---------------------------------------------------------------------------
// fp32 -> bf16 pre-conversion, 8 elements/thread (float4 x2 -> uint4).
// ---------------------------------------------------------------------------
__global__ __launch_bounds__(256) void cvt_kernel(const float* __restrict__ in,
                                                  unsigned short* __restrict__ out,
                                                  int n8) {
  const int i = blockIdx.x * 256 + threadIdx.x;
  if (i >= n8) return;
  const float4* p = (const float4*)in + (size_t)i * 2;
  const float4 a = p[0], b = p[1];
  union { unsigned short s[8]; uint4 u; } r;
  r.s[0] = f2bf(a.x); r.s[1] = f2bf(a.y); r.s[2] = f2bf(a.z); r.s[3] = f2bf(a.w);
  r.s[4] = f2bf(b.x); r.s[5] = f2bf(b.y); r.s[6] = f2bf(b.z); r.s[7] = f2bf(b.w);
  ((uint4*)out)[i] = r.u;
}

// ---------------------------------------------------------------------------
// bf16 GEMM: Y[row,col] = act( sum_k A[row,k]*W[col,k] + bias[col] )
// 128x128 macro-tile, 128 threads = 4 wave32 in a 2x2 grid, 64x64 per wave.
// Double-buffered LDS staged with async global->LDS b128 copies (ASYNCcnt).
// ---------------------------------------------------------------------------
template <int ACT, int OUTM>
__global__ __launch_bounds__(128) void gemm_bf16_kernel(
    const unsigned short* __restrict__ A, const unsigned short* __restrict__ W,
    const float* __restrict__ bias, void* __restrict__ Out) {
  __shared__ unsigned short As[2][128 * 32];
  __shared__ unsigned short Ws[2][128 * 32];

  const int tid  = threadIdx.x;
  const int wave = tid >> 5;
  const int lane = tid & 31;
  const int half = lane >> 4;
  const int l15  = lane & 15;
  const int row0 = blockIdx.x * 128;
  const int col0 = blockIdx.y * 128;
  const int wr   = (wave >> 1) * 64;  // wave row offset in tile
  const int wc   = (wave & 1) * 64;   // wave col offset in tile

  const unsigned short* ga = A + (size_t)(row0 + tid) * DMODEL;  // thread stages row tid
  const unsigned short* gw = W + (size_t)(col0 + tid) * DMODEL;

  v8f acc[16];
#pragma unroll
  for (int i = 0; i < 16; ++i) acc[i] = (v8f)0.0f;

  auto stage = [&](int buf, int k0) {
#pragma unroll
    for (int q = 0; q < 4; ++q) {
      async_copy_b128((unsigned)(size_t)&As[buf][tid * 32 + q * 8], ga + k0 + q * 8);
      async_copy_b128((unsigned)(size_t)&Ws[buf][tid * 32 + q * 8], gw + k0 + q * 8);
    }
  };

  stage(0, 0);
  for (int s = 0; s < DMODEL / 32; ++s) {
    const int buf = s & 1;
    const bool more = (s + 1) < (DMODEL / 32);
    if (more) {
      stage(buf ^ 1, (s + 1) * 32);  // prefetch next K-tile (async, 8 ops in flight)
      WAIT_ASYNC(8);                 // current buffer's 8 copies complete (in-order)
    } else {
      WAIT_ASYNC(0);
    }
    __syncthreads();

    Frag a[4];
#pragma unroll
    for (int mt = 0; mt < 4; ++mt) {
      const uint4* ar = (const uint4*)&As[buf][(wr + 16 * mt + l15) * 32];
      a[mt].u[0] = ar[2 * half];      // K = 16*half + 0..7
      a[mt].u[1] = ar[2 * half + 1];  // K = 16*half + 8..15
    }
#pragma unroll
    for (int nt = 0; nt < 4; ++nt) {
      Frag b;
      const uint4* br = (const uint4*)&Ws[buf][(wc + 16 * nt + l15) * 32];
      b.u[0] = br[2 * half];      // K = 16*half + 0..7
      b.u[1] = br[2 * half + 1];  // K = 16*half + 8..15
#pragma unroll
      for (int mt = 0; mt < 4; ++mt) acc[mt * 4 + nt] = wmma_bf16(a[mt], b, acc[mt * 4 + nt]);
    }
    __syncthreads();
  }

  // epilogue
#pragma unroll
  for (int nt = 0; nt < 4; ++nt) {
    const int col = col0 + wc + 16 * nt + l15;
    const float bv = bias[col];
#pragma unroll
    for (int mt = 0; mt < 4; ++mt) {
#pragma unroll
      for (int g = 0; g < 8; ++g) {
        const int row = row0 + wr + 16 * mt + g + 8 * half;
        float v = acc[mt * 4 + nt][g] + bv;
        if constexpr (ACT == ACT_PHI) v = (v > 0.f) ? (v + 1.f) : __expf(v);
        if constexpr (OUTM == OUT_FINAL) {
          ((float*)Out)[(size_t)row * DMODEL + col] = v;
        } else {
          const unsigned short hv = f2bf(v);
          const int b = row >> 12, n = row & (SEQ - 1);
          const int h = col >> 6, d = col & (HDIM - 1);
          unsigned short* o = (unsigned short*)Out;
          if constexpr (OUTM == OUT_Q)
            o[((size_t)(b * NHEADS + h) * SEQ + n) * HDIM + d] = hv;   // [B,H,N,D]
          else
            o[((size_t)(b * NHEADS + h) * HDIM + d) * SEQ + n] = hv;  // [B,H,D,M]
        }
      }
    }
  }
}

// ---------------------------------------------------------------------------
// kv[d,e] = sum_m K[m,d]*V[m,e]; z[d] = sum_m K[m,d] computed with an extra
// WMMA against an all-ones B matrix (every column of that tile equals z).
// K,V supplied transposed [B,H,64,M]; fragments are contiguous b128 loads.
// ---------------------------------------------------------------------------
__global__ __launch_bounds__(128) void kv_kernel(
    const unsigned short* __restrict__ Kt, const unsigned short* __restrict__ Vt,
    unsigned short* __restrict__ kvT, float* __restrict__ Z) {
  const int bh   = blockIdx.x;
  const int tid  = threadIdx.x;
  const int wave = tid >> 5;
  const int lane = tid & 31;
  const int half = lane >> 4;
  const int l15  = lane & 15;

  const unsigned short* arow = Kt + ((size_t)bh * HDIM + 16 * wave + l15) * SEQ;
  const unsigned short* brow[4];
#pragma unroll
  for (int nt = 0; nt < 4; ++nt)
    brow[nt] = Vt + ((size_t)bh * HDIM + 16 * nt + l15) * SEQ;

  Frag ones;
  ones.u[0].x = ones.u[0].y = ones.u[0].z = ones.u[0].w = 0x3F803F80u;  // bf16 1.0 x2
  ones.u[1] = ones.u[0];

  v8f acc[4], accz;
#pragma unroll
  for (int i = 0; i < 4; ++i) acc[i] = (v8f)0.0f;
  accz = (v8f)0.0f;

  for (int m0 = 0; m0 < SEQ; m0 += 32) {
    Frag a;
    const uint4* ap = (const uint4*)(arow + m0);
    a.u[0] = ap[2 * half];
    a.u[1] = ap[2 * half + 1];
#pragma unroll
    for (int nt = 0; nt < 4; ++nt) {
      Frag b;
      const uint4* bp = (const uint4*)(brow[nt] + m0);
      b.u[0] = bp[2 * half];
      b.u[1] = bp[2 * half + 1];
      acc[nt] = wmma_bf16(a, b, acc[nt]);
    }
    accz = wmma_bf16(a, ones, accz);  // z partials ride the matrix pipe
  }

#pragma unroll
  for (int nt = 0; nt < 4; ++nt)
#pragma unroll
    for (int g = 0; g < 8; ++g) {
      const int d = 16 * wave + g + 8 * half;
      const int e = 16 * nt + l15;
      kvT[(size_t)bh * (HDIM * HDIM) + e * HDIM + d] = f2bf(acc[nt][g]);  // kv^T
    }
  if (l15 == 0) {  // column 0 of the ones-accumulator holds z[d]
#pragma unroll
    for (int g = 0; g < 8; ++g) {
      const int d = 16 * wave + g + 8 * half;
      Z[(size_t)bh * HDIM + d] = accz[g];
    }
  }
}

// ---------------------------------------------------------------------------
// o1[n, h*64+e] = (q[n,:] @ kv[:,e]) / (q[n,:]@z + eps), bf16 out [B*N, 1024].
// ---------------------------------------------------------------------------
__global__ __launch_bounds__(128) void attn_kernel(
    const unsigned short* __restrict__ Q, const unsigned short* __restrict__ kvT,
    const float* __restrict__ Z, unsigned short* __restrict__ O1) {
  __shared__ float zs[HDIM];
  __shared__ float dens[64];
  const int bh   = blockIdx.y;
  const int n0   = blockIdx.x * 64;
  const int tid  = threadIdx.x;
  const int wave = tid >> 5;
  const int lane = tid & 31;
  const int half = lane >> 4;
  const int l15  = lane & 15;

  if (tid < HDIM) zs[tid] = Z[(size_t)bh * HDIM + tid];
  __syncthreads();
  if (tid < 64) {
    const unsigned short* qr = Q + ((size_t)bh * SEQ + n0 + tid) * HDIM;
    float s0 = 0.f, s1 = 0.f, s2 = 0.f, s3 = 0.f;
#pragma unroll 4
    for (int d = 0; d < HDIM; d += 4) {
      s0 += bf2f(qr[d + 0]) * zs[d + 0];
      s1 += bf2f(qr[d + 1]) * zs[d + 1];
      s2 += bf2f(qr[d + 2]) * zs[d + 2];
      s3 += bf2f(qr[d + 3]) * zs[d + 3];
    }
    dens[tid] = (s0 + s1) + (s2 + s3) + EPSV;
  }
  __syncthreads();

  v8f acc[4];
#pragma unroll
  for (int i = 0; i < 4; ++i) acc[i] = (v8f)0.0f;

  const unsigned short* arow = Q + ((size_t)bh * SEQ + n0 + 16 * wave + l15) * HDIM;
#pragma unroll
  for (int d0 = 0; d0 < HDIM; d0 += 32) {
    Frag a;
    const uint4* ap = (const uint4*)(arow + d0);
    a.u[0] = ap[2 * half];
    a.u[1] = ap[2 * half + 1];
#pragma unroll
    for (int nt = 0; nt < 4; ++nt) {
      Frag b;
      const uint4* bp =
          (const uint4*)(kvT + (size_t)bh * (HDIM * HDIM) + (16 * nt + l15) * HDIM + d0);
      b.u[0] = bp[2 * half];
      b.u[1] = bp[2 * half + 1];
      acc[nt] = wmma_bf16(a, b, acc[nt]);
    }
  }

  const int bb = bh >> 4, hh = bh & 15;
#pragma unroll
  for (int nt = 0; nt < 4; ++nt)
#pragma unroll
    for (int g = 0; g < 8; ++g) {
      const int rl = 16 * wave + g + 8 * half;
      const float v = acc[nt][g] / dens[rl];
      const int n = n0 + rl;
      const int col = hh * HDIM + 16 * nt + l15;
      O1[((size_t)bb * SEQ + n) * DMODEL + col] = f2bf(v);
    }
}

// ---------------------------------------------------------------------------
extern "C" void kernel_launch(void* const* d_in, const int* in_sizes, int n_in,
                              void* d_out, int out_size, void* d_ws, size_t ws_size,
                              hipStream_t stream) {
  const float* queries = (const float*)d_in[0];
  const float* keys    = (const float*)d_in[1];
  const float* values  = (const float*)d_in[2];
  const float* wq = (const float*)d_in[3];
  const float* bq = (const float*)d_in[4];
  const float* wk = (const float*)d_in[5];
  const float* bk = (const float*)d_in[6];
  const float* wv = (const float*)d_in[7];
  const float* bv = (const float*)d_in[8];
  const float* wo = (const float*)d_in[9];
  const float* bo = (const float*)d_in[10];

  char* ws = (char*)d_ws;
  const size_t SZ = (size_t)ROWS * DMODEL * sizeof(unsigned short);        // 32 MB
  const size_t WZ = (size_t)DMODEL * DMODEL * sizeof(unsigned short);      // 2 MB
  unsigned short* bufA  = (unsigned short*)(ws);                           // staging act
  unsigned short* wbuf  = (unsigned short*)(ws + SZ);                      // staging wgt
  unsigned short* q_ws  = (unsigned short*)(ws + SZ + WZ);
  unsigned short* kt_ws = (unsigned short*)(ws + 2 * SZ + WZ);
  unsigned short* vt_ws = (unsigned short*)(ws + 3 * SZ + WZ);
  unsigned short* o1_ws = (unsigned short*)(ws + 4 * SZ + WZ);
  unsigned short* kv_ws = (unsigned short*)(ws + 5 * SZ + WZ);
  float* z_ws = (float*)(ws + 5 * SZ + WZ + (size_t)BATCH * NHEADS * HDIM * HDIM * 2);

  const int nA8 = ROWS * DMODEL / 8, nW8 = DMODEL * DMODEL / 8;
  const dim3 cgA((nA8 + 255) / 256), cgW((nW8 + 255) / 256);
  const dim3 gg(ROWS / 128, DMODEL / 128);

  cvt_kernel<<<cgA, 256, 0, stream>>>(queries, bufA, nA8);
  cvt_kernel<<<cgW, 256, 0, stream>>>(wq, wbuf, nW8);
  gemm_bf16_kernel<ACT_PHI, OUT_Q><<<gg, 128, 0, stream>>>(bufA, wbuf, bq, q_ws);

  cvt_kernel<<<cgA, 256, 0, stream>>>(keys, bufA, nA8);
  cvt_kernel<<<cgW, 256, 0, stream>>>(wk, wbuf, nW8);
  gemm_bf16_kernel<ACT_PHI, OUT_TR><<<gg, 128, 0, stream>>>(bufA, wbuf, bk, kt_ws);

  cvt_kernel<<<cgA, 256, 0, stream>>>(values, bufA, nA8);
  cvt_kernel<<<cgW, 256, 0, stream>>>(wv, wbuf, nW8);
  gemm_bf16_kernel<ACT_NONE, OUT_TR><<<gg, 128, 0, stream>>>(bufA, wbuf, bv, vt_ws);

  kv_kernel<<<dim3(BATCH * NHEADS), 128, 0, stream>>>(kt_ws, vt_ws, kv_ws, z_ws);
  attn_kernel<<<dim3(SEQ / 64, BATCH * NHEADS), 128, 0, stream>>>(q_ws, kv_ws, z_ws, o1_ws);

  cvt_kernel<<<cgW, 256, 0, stream>>>(wo, wbuf, nW8);
  gemm_bf16_kernel<ACT_NONE, OUT_FINAL><<<gg, 128, 0, stream>>>(o1_ws, wbuf, bo, d_out);
}
